// LongHorizonCuriosityLoss_52295521796894
// MI455X (gfx1250) — compile-verified, matrix operand order checked by reading, not run on hardware
//
#include <hip/hip_runtime.h>

// Problem constants (from reference): T=64, B=1024, D=1024
#define T_STEPS 64
#define BATCH   1024
#define DIM     1024
#define ROWS    16           // batch rows per workgroup
#define NTHREADS 256         // 8 waves (wave32)
#define NWAVES   8
#define NT_PER_WAVE 8        // 8 x 16 = 128 output cols per wave

typedef __attribute__((ext_vector_type(16))) __bf16 v16bf;
typedef __attribute__((ext_vector_type(8)))  __bf16 v8bf;
typedef __attribute__((ext_vector_type(4)))  __bf16 v4bf;
typedef __attribute__((ext_vector_type(8)))  float  v8f;

union ABfrag { v16bf v; v8bf h[2]; };

// ---------------------------------------------------------------------------
// Pre-kernel: convert W (f32, row-major [D][D]) to bf16 in workspace.
// W.bf16 is 2 MB -> resides in the 192 MB L2 for the whole scan.
// ---------------------------------------------------------------------------
__global__ void __launch_bounds__(256)
wconvert_kernel(const float* __restrict__ W, __bf16* __restrict__ Wb)
{
    int i = (blockIdx.x * 256 + threadIdx.x) * 4;
    float4 w4 = *(const float4*)(W + i);
    v4bf o;
    o[0] = (__bf16)w4.x; o[1] = (__bf16)w4.y;
    o[2] = (__bf16)w4.z; o[3] = (__bf16)w4.w;
    *(v4bf*)(Wb + i) = o;
}

// ---------------------------------------------------------------------------
// Persistent scan kernel: one workgroup owns 16 batch rows for all 64 steps.
// State (loss, weight) lives in LDS; se is staged in LDS as bf16 for WMMA.
// ---------------------------------------------------------------------------
__global__ void __launch_bounds__(NTHREADS)
curiosity_scan_kernel(const float* __restrict__ pred,
                      const float* __restrict__ tru,
                      const __bf16* __restrict__ Wb,
                      const float* __restrict__ bias,
                      float* __restrict__ out)
{
    __shared__ float  s_loss[ROWS][DIM];   // 64 KB : accumulated loss
    __shared__ float  s_wl[ROWS][DIM];     // 64 KB : weight -> logits -> weight'
    __shared__ __bf16 s_se[ROWS][DIM];     // 32 KB : se tile in bf16 (WMMA A)

    const int tid  = threadIdx.x;
    const int lane = tid & 31;
    const int wave = tid >> 5;
    const int r    = tid >> 4;     // 0..15 : row for elementwise / softmax work
    const int c16  = tid & 15;     // 16 threads cooperate per row
    const int rb   = blockIdx.x * ROWS;

    // WMMA lane geometry (16-bit A/B layout per CDNA5 ISA 7.12.2):
    // lanes 0..15 hold K = k0..k0+7 (+16..+23 in upper VGPRs),
    // lanes 16..31 hold K = k0+8..k0+15 (+24..+31).
    const int m      = lane & 15;                 // A-row / B-col within tile
    const int khalf  = (lane >> 4) ? 8 : 0;
    const int n_base = wave * (16 * NT_PER_WAVE) + m;

    // ---- init scan carry: loss = 0, weight = 1 ----
    for (int i = tid; i < ROWS * DIM; i += NTHREADS) {
        (&s_loss[0][0])[i] = 0.0f;
        (&s_wl[0][0])[i]   = 1.0f;
    }
    __syncthreads();

    #pragma unroll 1
    for (int t = 0; t < T_STEPS; ++t) {
        const float* pr = pred + ((size_t)t * BATCH + rb + r) * DIM;
        const float* tr = tru  + ((size_t)t * BATCH + rb + r) * DIM;

        // Prefetch next step's rows (global_prefetch_b8) to hide HBM latency
        // behind this step's WMMA work.
        if (t + 1 < T_STEPS) {
            __builtin_prefetch(pr + (size_t)BATCH * DIM, 0, 1);
            __builtin_prefetch(tr + (size_t)BATCH * DIM, 0, 1);
        }

        // ---- stage 1: se = (pred-true)^2 ; loss += weight*se ; stash se ----
        #pragma unroll
        for (int i = 0; i < DIM / 64; ++i) {
            const int col = (c16 << 2) + (i << 6);
            float4 p4 = *(const float4*)(pr + col);
            float4 q4 = *(const float4*)(tr + col);
            float dx = p4.x - q4.x, dy = p4.y - q4.y;
            float dz = p4.z - q4.z, dw = p4.w - q4.w;
            float sx = dx * dx, sy = dy * dy, sz = dz * dz, sw = dw * dw;

            float4 w4 = *(const float4*)&s_wl[r][col];
            float4 l4 = *(const float4*)&s_loss[r][col];
            l4.x = fmaf(w4.x, sx, l4.x);
            l4.y = fmaf(w4.y, sy, l4.y);
            l4.z = fmaf(w4.z, sz, l4.z);
            l4.w = fmaf(w4.w, sw, l4.w);
            *(float4*)&s_loss[r][col] = l4;

            v4bf sb;
            sb[0] = (__bf16)sx; sb[1] = (__bf16)sy;
            sb[2] = (__bf16)sz; sb[3] = (__bf16)sw;
            *(v4bf*)&s_se[r][col] = sb;
        }
        __syncthreads();   // se complete; weight reads done before overwrite

        // ---- stage 2: logits = se @ W^T + b via v_wmma_f32_16x16x32_bf16 ----
        v8f acc[NT_PER_WAVE];
        const v8f vzero = {0.f, 0.f, 0.f, 0.f, 0.f, 0.f, 0.f, 0.f};
        #pragma unroll
        for (int tt = 0; tt < NT_PER_WAVE; ++tt) acc[tt] = vzero;

        #pragma unroll 1
        for (int kt = 0; kt < DIM / 32; ++kt) {
            const int k0 = kt * 32 + khalf;
            ABfrag a;
            a.h[0] = *(const v8bf*)&s_se[m][k0];        // K k0..k0+7
            a.h[1] = *(const v8bf*)&s_se[m][k0 + 16];   // K k0+16..k0+23
            #pragma unroll
            for (int tt = 0; tt < NT_PER_WAVE; ++tt) {
                // B[k][n] = W[n][k]: lane's column n is a row of W -> two
                // contiguous 16-byte loads from L2-resident bf16 W.
                const __bf16* wrow = Wb + (size_t)(n_base + tt * 16) * DIM;
                ABfrag b;
                b.h[0] = *(const v8bf*)(wrow + k0);
                b.h[1] = *(const v8bf*)(wrow + k0 + 16);
                acc[tt] = __builtin_amdgcn_wmma_f32_16x16x32_bf16(
                    false, a.v, false, b.v, (short)0, acc[tt], false, false);
            }
        }

        // write logits + bias into the (now free) weight buffer
        #pragma unroll
        for (int tt = 0; tt < NT_PER_WAVE; ++tt) {
            const int n  = n_base + tt * 16;
            const float bv = bias[n];
            #pragma unroll
            for (int v = 0; v < 8; ++v) {
                const int mrow = (lane < 16) ? v : (v + 8);  // C/D layout
                s_wl[mrow][n] = acc[tt][v] + bv;
            }
        }
        __syncthreads();

        // ---- stage 3: row softmax; weight' = se * softmax(logits) ----
        // 16 lanes per row (one wave32 half) -> shfl_xor reductions.
        float mx = -3.402823466e+38f;
        #pragma unroll
        for (int i = 0; i < DIM / 16; ++i)
            mx = fmaxf(mx, s_wl[r][c16 + (i << 4)]);
        #pragma unroll
        for (int off = 8; off >= 1; off >>= 1)
            mx = fmaxf(mx, __shfl_xor(mx, off, 32));

        float ssum = 0.0f;
        #pragma unroll
        for (int i = 0; i < DIM / 16; ++i) {
            const int col = c16 + (i << 4);
            const float e = __expf(s_wl[r][col] - mx);
            s_wl[r][col] = e;                 // own column only
            ssum += e;
        }
        #pragma unroll
        for (int off = 8; off >= 1; off >>= 1)
            ssum += __shfl_xor(ssum, off, 32);
        const float inv = 1.0f / ssum;

        #pragma unroll
        for (int i = 0; i < DIM / 16; ++i) {
            const int col = c16 + (i << 4);
            s_wl[r][col] = (float)s_se[r][col] * s_wl[r][col] * inv;
        }
        __syncthreads();   // weight' ready before next step reads it
    }

    // ---- emit accumulated loss ----
    #pragma unroll
    for (int i = 0; i < DIM / 64; ++i) {
        const int col = (c16 << 2) + (i << 6);
        *(float4*)(out + (size_t)(rb + r) * DIM + col) =
            *(const float4*)&s_loss[r][col];
    }
}

// ---------------------------------------------------------------------------
extern "C" void kernel_launch(void* const* d_in, const int* in_sizes, int n_in,
                              void* d_out, int out_size, void* d_ws, size_t ws_size,
                              hipStream_t stream) {
    const float* pred = (const float*)d_in[0];
    const float* tru  = (const float*)d_in[1];
    const float* W    = (const float*)d_in[2];
    const float* bias = (const float*)d_in[3];
    float*       out  = (float*)d_out;
    __bf16*      Wb   = (__bf16*)d_ws;      // 1024*1024 bf16 = 2 MB scratch

    wconvert_kernel<<<(DIM * DIM) / (256 * 4), 256, 0, stream>>>(W, Wb);
    curiosity_scan_kernel<<<BATCH / ROWS, NTHREADS, 0, stream>>>(
        pred, tru, Wb, bias, out);
}
